// PentachoronCrossAttention_28509992910915
// MI455X (gfx1250) — matrix operand order repaired and checked by reading, not compiled, not run on hardware
//
#include <hip/hip_runtime.h>

// ---------------------------------------------------------------------------
// PentachoronCrossAttention for MI455X (gfx1250, wave32, WMMA + TDM)
//   Stage 0: convert vertices / in_proj_w / out_proj_w fp32 -> bf16 (once)
//   Stage 1: QKV GEMM   [40960,1792] x [5376,1792]^T -> bf16 qkv (ws)
//   Stage 2: masked 5x5 attention per (b,h)          -> bf16 attn (ws)
//   Stage 3: out GEMM   [40960,1792] x [1792,1792]^T -> fp32 d_out
// GEMMs: v_wmma_f32_16x16x32_bf16, double-buffered LDS fed by
// tensor_load_to_lds (TDM) with s_wait_tensorcnt pipelining.
// ---------------------------------------------------------------------------

typedef __attribute__((ext_vector_type(16))) __bf16 v16bf;
typedef __attribute__((ext_vector_type(8)))  __bf16 v8bf;
typedef __attribute__((ext_vector_type(8)))  float  v8f;
typedef __attribute__((ext_vector_type(4)))  unsigned int u32x4;
typedef __attribute__((ext_vector_type(8)))  int i32x8;
typedef __attribute__((ext_vector_type(4)))  int i32x4;

#define BM 128
#define BN 128
#define BK 64
#define LDSK 72   // BK + 8: 144-byte rows -> 16B aligned, bank-skewed

#define USE_TDM (__has_builtin(__builtin_amdgcn_tensor_load_to_lds) && \
                 __has_builtin(__builtin_amdgcn_s_wait_tensorcnt))

__device__ __forceinline__ unsigned short f2bf(float f) {
  union { float f; unsigned int u; } c; c.f = f;
  unsigned int u = c.u;
  u += 0x7fffu + ((u >> 16) & 1u);        // round-to-nearest-even
  return (unsigned short)(u >> 16);
}
__device__ __forceinline__ float bf2f(unsigned short h) {
  union { float f; unsigned int u; } c; c.u = ((unsigned int)h) << 16;
  return c.f;
}

#if USE_TDM
// Issue one TDM load of a [tile_rows x BK] bf16 tile (row stride k_stride
// elements) into LDS at lds_off, padding each 128B row by 16B (-> LDSK rows).
// D# layout per CDNA5 ISA 8.3/8.4 (2D tensor, groups 2/3 NULL-equivalent zeros).
__device__ __forceinline__ void tdm_load_tile_bf16(
    const unsigned short* gtile, unsigned lds_off,
    unsigned k_stride, unsigned dim1_len, unsigned tile_rows)
{
  unsigned long long ga = (unsigned long long)(size_t)gtile;
  u32x4 g0 = { 1u,                                    // count=1, user mode
               lds_off,                               // lds_addr
               (unsigned)ga,                          // global_addr[31:0]
               (unsigned)((ga >> 32) & 0x01ffffffu) | (2u << 30) }; // [56:32]+type=2
  i32x8 g1;
  g1[0] = (int)((1u << 16)        // data_size = 2 bytes
              | (1u << 20)        // pad_enable
              | (4u << 22)        // pad_interval: 32 DWORDs (=128B = BK bf16)
              | (3u << 25));      // pad_amount: 4 DWORDs (=16B = 8 bf16)
  g1[1] = (int)((k_stride & 0xffffu) << 16);                    // tensor_dim0 lo16
  g1[2] = (int)((k_stride >> 16) | ((dim1_len & 0xffffu) << 16)); // dim0 hi / dim1 lo
  g1[3] = (int)((dim1_len >> 16) | ((unsigned)BK << 16));        // dim1 hi / tile_dim0
  g1[4] = (int)tile_rows;                                        // tile_dim1, tile_dim2=0
  g1[5] = (int)k_stride;                                         // tensor_dim0_stride lo32
  g1[6] = 0;                                                     // stride hi / dim1_stride lo
  g1[7] = 0;
  i32x4 z4 = {0, 0, 0, 0};
#if defined(__clang_major__) && (__clang_major__ >= 23)
  i32x8 z8 = {0, 0, 0, 0, 0, 0, 0, 0};
  __builtin_amdgcn_tensor_load_to_lds(g0, g1, z4, z4, z8, 0);
#else
  __builtin_amdgcn_tensor_load_to_lds(g0, g1, z4, z4, 0);
#endif
}
#endif

// ---------------------------------------------------------------------------
// fp32 -> bf16 elementwise convert (float4 per thread)
// ---------------------------------------------------------------------------
__global__ __launch_bounds__(256) void cvt_f32_bf16(
    const float* __restrict__ in, unsigned short* __restrict__ out, int n4)
{
  int i = blockIdx.x * 256 + threadIdx.x;
  if (i < n4) {
    float4 v = ((const float4*)in)[i];
    ushort4 p = { f2bf(v.x), f2bf(v.y), f2bf(v.z), f2bf(v.w) };
    ((ushort4*)out)[i] = p;
  }
}

// ---------------------------------------------------------------------------
// C[M,N] = A[M,K] * W[N,K]^T + bias[N];  A,W bf16 row-major K-contig.
// OUT_BF16 selects bf16 vs fp32 C.
// Grid: (N/BN, M/BM) so consecutive blocks share the A panel (L2 reuse).
// ---------------------------------------------------------------------------
template <bool OUT_BF16>
__global__ __launch_bounds__(256) void gemm_wmma_bf16(
    const unsigned short* __restrict__ A, const unsigned short* __restrict__ W,
    const float* __restrict__ bias, void* __restrict__ Cptr,
    int M, int N, int K)
{
  __shared__ __align__(16) unsigned short As[2][BM][LDSK];
  __shared__ __align__(16) unsigned short Bs[2][BN][LDSK];

  const int tid  = threadIdx.x;
  const int lane = tid & 31;
  const int wave = tid >> 5;
  const int wm   = wave & 3;    // wave M offset: wm*32  (2 tiles)
  const int wn   = wave >> 2;   // wave N offset: wn*64  (4 tiles)
  const int n0   = blockIdx.x * BN;
  const int m0   = blockIdx.y * BM;
  const int NT   = K / BK;

  v8f acc[2][4];
#pragma unroll
  for (int mi = 0; mi < 2; ++mi)
#pragma unroll
    for (int ni = 0; ni < 4; ++ni)
      acc[mi][ni] = (v8f){0.f,0.f,0.f,0.f,0.f,0.f,0.f,0.f};

#if USE_TDM
  const unsigned as_off = (unsigned)(size_t)&As[0][0][0];
  const unsigned bs_off = (unsigned)(size_t)&Bs[0][0][0];
  const unsigned abuf_sz = (unsigned)(BM * LDSK * 2);
  // prologue: DMA tile 0 into buffer 0 (wave0: A panel, wave1: B panel)
  if (wave == 0)
    tdm_load_tile_bf16(A + (size_t)m0 * K, as_off, (unsigned)K, (unsigned)M, BM);
  else if (wave == 1)
    tdm_load_tile_bf16(W + (size_t)n0 * K, bs_off, (unsigned)K, (unsigned)N, BN);
#endif

  for (int kt = 0; kt < NT; ++kt) {
#if USE_TDM
    const int cur = kt & 1;
    if (kt + 1 < NT) {
      // DMA next tile into the other buffer (free since end-barrier of kt-1),
      // then wait for the *current* tile only (in-order per-wave completion).
      const int k1 = (kt + 1) * BK;
      if (wave == 0)
        tdm_load_tile_bf16(A + (size_t)m0 * K + k1,
                           as_off + (unsigned)(cur ^ 1) * abuf_sz,
                           (unsigned)K, (unsigned)M, BM);
      else if (wave == 1)
        tdm_load_tile_bf16(W + (size_t)n0 * K + k1,
                           bs_off + (unsigned)(cur ^ 1) * abuf_sz,
                           (unsigned)K, (unsigned)N, BN);
      __builtin_amdgcn_s_wait_tensorcnt(1);
    } else {
      __builtin_amdgcn_s_wait_tensorcnt(0);
    }
    __syncthreads();
#else
    const int cur = 0;
    __syncthreads();   // previous iteration's readers done before overwrite
    {
      const int k1 = kt * BK;
#pragma unroll
      for (int i = 0; i < 4; ++i) {               // A: 1024 x 16B chunks
        int f = tid + i * 256;
        int row = f >> 3, c8 = f & 7;
        *(uint4*)&As[0][row][c8 * 8] =
            *(const uint4*)&A[(size_t)(m0 + row) * K + k1 + c8 * 8];
      }
#pragma unroll
      for (int i = 0; i < 4; ++i) {               // B
        int f = tid + i * 256;
        int row = f >> 3, c8 = f & 7;
        *(uint4*)&Bs[0][row][c8 * 8] =
            *(const uint4*)&W[(size_t)(n0 + row) * K + k1 + c8 * 8];
      }
    }
    __syncthreads();
#endif

    // ---- 16 WMMAs from buf[cur]: 2 k-slices of 32 x (2 m-tiles x 4 n-tiles)
    const int khalf = (lane >> 4) << 3;   // A: 0 or 8
    const int kb    = (lane >> 4) << 4;   // 0 or 16; unused, B also uses khalf
#pragma unroll
    for (int s = 0; s < 2; ++s) {
      const int sk = s * 32;
      v16bf af[2];
#pragma unroll
      for (int mi = 0; mi < 2; ++mi) {
        int row = wm * 32 + mi * 16 + (lane & 15);
        v8bf lo = *(const v8bf*)&As[cur][row][sk + khalf];
        v8bf hi = *(const v8bf*)&As[cur][row][sk + 16 + khalf];
#pragma unroll
        for (int t = 0; t < 8; ++t) { af[mi][t] = lo[t]; af[mi][t + 8] = hi[t]; }
      }
      v16bf bfm[4];
#pragma unroll
      for (int ni = 0; ni < 4; ++ni) {
        int row = wn * 64 + ni * 16 + (lane & 15);
        v8bf lo = *(const v8bf*)&Bs[cur][row][sk + khalf];
        v8bf hi = *(const v8bf*)&Bs[cur][row][sk + 16 + khalf];
#pragma unroll
        for (int t = 0; t < 8; ++t) { bfm[ni][t] = lo[t]; bfm[ni][t + 8] = hi[t]; }
      }
#pragma unroll
      for (int mi = 0; mi < 2; ++mi)
#pragma unroll
        for (int ni = 0; ni < 4; ++ni)
          acc[mi][ni] = __builtin_amdgcn_wmma_f32_16x16x32_bf16(
              false, af[mi], false, bfm[ni], (short)0, acc[mi][ni], false, false);
    }
    __syncthreads();
  }

  // ---- epilogue: bias + store (C/D layout: n = lane&15, VGPR j -> m += j)
#pragma unroll
  for (int mi = 0; mi < 2; ++mi) {
#pragma unroll
    for (int ni = 0; ni < 4; ++ni) {
      int n     = n0 + wn * 64 + ni * 16 + (lane & 15);
      float bv  = bias[n];
      int mbase = m0 + wm * 32 + mi * 16 + ((lane >> 4) << 3);
#pragma unroll
      for (int j = 0; j < 8; ++j) {
        float val = acc[mi][ni][j] + bv;
        if constexpr (OUT_BF16)
          ((unsigned short*)Cptr)[(size_t)(mbase + j) * N + n] = f2bf(val);
        else
          ((float*)Cptr)[(size_t)(mbase + j) * N + n] = val;
      }
    }
  }
}

// ---------------------------------------------------------------------------
// Masked 5x5 attention. One wave per (b,h). Lane l owns d = 4l..4l+3.
// ---------------------------------------------------------------------------
__global__ __launch_bounds__(256) void attn_5x5_kernel(
    const unsigned short* __restrict__ qkv,  // [B*5, 5376] bf16
    const int* __restrict__ adj,             // [5,5]
    unsigned short* __restrict__ outp)       // [B*5, 1792] bf16
{
  const int lane = threadIdx.x & 31;
  const int wave = threadIdx.x >> 5;
  const int gidx = blockIdx.x * 8 + wave;    // (b,h)
  const int b = gidx / 14;
  const int h = gidx % 14;

  const size_t rs = 5376;
  const size_t base = (size_t)b * 5 * rs + (size_t)h * 128 + (size_t)lane * 4;

  float q[5][4], k[5][4], v[5][4];
#pragma unroll
  for (int i = 0; i < 5; ++i) {
    ushort4 qq = *(const ushort4*)&qkv[base + (size_t)i * rs];
    ushort4 kk = *(const ushort4*)&qkv[base + (size_t)i * rs + 1792];
    ushort4 vv = *(const ushort4*)&qkv[base + (size_t)i * rs + 3584];
    q[i][0] = bf2f(qq.x); q[i][1] = bf2f(qq.y); q[i][2] = bf2f(qq.z); q[i][3] = bf2f(qq.w);
    k[i][0] = bf2f(kk.x); k[i][1] = bf2f(kk.y); k[i][2] = bf2f(kk.z); k[i][3] = bf2f(kk.w);
    v[i][0] = bf2f(vv.x); v[i][1] = bf2f(vv.y); v[i][2] = bf2f(vv.z); v[i][3] = bf2f(vv.w);
  }

  float s[5][5];
#pragma unroll
  for (int i = 0; i < 5; ++i)
#pragma unroll
    for (int j = 0; j < 5; ++j) {
      float p = q[i][0]*k[j][0] + q[i][1]*k[j][1] + q[i][2]*k[j][2] + q[i][3]*k[j][3];
#pragma unroll
      for (int off = 16; off > 0; off >>= 1)
        p += __shfl_xor(p, off, 32);
      s[i][j] = p;
    }

  const float scale = 0.08838834764831843f;  // 1/sqrt(128)
#pragma unroll
  for (int i = 0; i < 5; ++i) {
    float mx = -3.4e38f;
#pragma unroll
    for (int j = 0; j < 5; ++j) {
      float m = (adj[i * 5 + j] == 0) ? -1e30f : 0.0f;
      s[i][j] = s[i][j] * scale + m;
      mx = fmaxf(mx, s[i][j]);
    }
    float den = 0.f;
#pragma unroll
    for (int j = 0; j < 5; ++j) { float e = __expf(s[i][j] - mx); s[i][j] = e; den += e; }
    float inv = 1.f / den;
#pragma unroll
    for (int j = 0; j < 5; ++j) s[i][j] *= inv;
  }

#pragma unroll
  for (int i = 0; i < 5; ++i) {
    float o0 = 0.f, o1 = 0.f, o2 = 0.f, o3 = 0.f;
#pragma unroll
    for (int j = 0; j < 5; ++j) {
      o0 += s[i][j] * v[j][0]; o1 += s[i][j] * v[j][1];
      o2 += s[i][j] * v[j][2]; o3 += s[i][j] * v[j][3];
    }
    ushort4 st = { f2bf(o0), f2bf(o1), f2bf(o2), f2bf(o3) };
    *(ushort4*)&outp[(size_t)(b * 5 + i) * 1792 + (size_t)h * 128 + (size_t)lane * 4] = st;
  }
}

// ---------------------------------------------------------------------------
extern "C" void kernel_launch(void* const* d_in, const int* in_sizes, int n_in,
                              void* d_out, int out_size, void* d_ws, size_t ws_size,
                              hipStream_t stream) {
  const float* vertices   = (const float*)d_in[0];  // [8192,5,1792]
  const int*   adjacency  = (const int*)d_in[1];    // [5,5]
  const float* in_proj_w  = (const float*)d_in[2];  // [5376,1792]
  const float* in_proj_b  = (const float*)d_in[3];  // [5376]
  const float* out_proj_w = (const float*)d_in[4];  // [1792,1792]
  const float* out_proj_b = (const float*)d_in[5];  // [1792]

  const int M = 8192 * 5;   // 40960
  const int D = 1792;
  const int E = 3 * D;      // 5376

  unsigned short* xb   = (unsigned short*)d_ws;        // M*D bf16
  unsigned short* w1b  = xb  + (size_t)M * D;          // E*D bf16
  unsigned short* w2b  = w1b + (size_t)E * D;          // D*D bf16
  unsigned short* qkv  = w2b + (size_t)D * D;          // M*E bf16
  unsigned short* attn = qkv + (size_t)M * E;          // M*D bf16

  dim3 blk(256);

  // Stage 0: one-time fp32 -> bf16 conversions
  {
    int n4;
    n4 = (M * D) / 4;
    cvt_f32_bf16<<<dim3((n4 + 255) / 256), blk, 0, stream>>>(vertices, xb, n4);
    n4 = (E * D) / 4;
    cvt_f32_bf16<<<dim3((n4 + 255) / 256), blk, 0, stream>>>(in_proj_w, w1b, n4);
    n4 = (D * D) / 4;
    cvt_f32_bf16<<<dim3((n4 + 255) / 256), blk, 0, stream>>>(out_proj_w, w2b, n4);
  }

  // Stage 1: QKV projection -> bf16
  gemm_wmma_bf16<true><<<dim3(E / BN, M / BM), blk, 0, stream>>>(
      xb, w1b, in_proj_b, qkv, M, E, D);

  // Stage 2: masked attention (B*H = 114688 waves, 8 per block)
  attn_5x5_kernel<<<dim3(8192 * 14 / 8), blk, 0, stream>>>(qkv, adjacency, attn);

  // Stage 3: output projection -> fp32 d_out
  gemm_wmma_bf16<false><<<dim3(D / BN, M / BM), blk, 0, stream>>>(
      attn, w2b, out_proj_b, d_out, M, D, D);
}
